// ResMambaLayer_59863254172044
// MI455X (gfx1250) — matrix operand order repaired and checked by reading, not compiled, not run on hardware
//
#include <hip/hip_runtime.h>
#include <hip/hip_bf16.h>
#include <math.h>

// ---------------- problem constants (from reference) ----------------
#define BSZ     2
#define CDIM    128
#define LSEQ    13824          // 24*24*24
#define MROWS   (BSZ * LSEQ)   // 27648 rows, multiple of 16
#define DINNER  256
#define NSTATE  16
#define DTRANK  8
#define XDBL_LD 48             // 40 padded to 48 (multiple of 16)

typedef __attribute__((ext_vector_type(16))) __bf16 v16bf;
typedef __attribute__((ext_vector_type(8)))  __bf16 v8bf;
typedef __attribute__((ext_vector_type(8)))  float  v8f;

// ---------------- block-wide sum over 128 threads ----------------
__device__ __forceinline__ float block_sum128(float v, float* s) {
    int t = threadIdx.x;
    s[t] = v;
    __syncthreads();
    #pragma unroll
    for (int off = 64; off > 0; off >>= 1) {
        if (t < off) s[t] += s[t + off];
        __syncthreads();
    }
    float r = s[0];
    __syncthreads();
    return r;
}

// ---------------- f32 -> bf16 elementwise (weights) ----------------
__global__ void f32_to_bf16_kernel(const float* __restrict__ in,
                                   __bf16* __restrict__ out, int n) {
    int t = blockIdx.x * blockDim.x + threadIdx.x;
    if (t < n) out[t] = (__bf16)in[t];
}

// ---------------- kernel 1: transpose (B,C,L)->(B,L,C) + LayerNorm -> bf16 ------
__global__ void transpose_ln_kernel(const float* __restrict__ x,
                                    const float* __restrict__ w,
                                    const float* __restrict__ bia,
                                    float* __restrict__ xf,
                                    __bf16* __restrict__ h1b) {
    __shared__ float s[128];
    int row = blockIdx.x;            // b*L + l
    int c   = threadIdx.x;           // 0..127
    int b = row / LSEQ, l = row - b * LSEQ;
    float v = x[((size_t)b * CDIM + c) * LSEQ + l];
    xf[(size_t)row * CDIM + c] = v;
    float mean = block_sum128(v, s) * (1.0f / 128.0f);
    float d = v - mean;
    float var = block_sum128(d * d, s) * (1.0f / 128.0f);
    h1b[(size_t)row * CDIM + c] = (__bf16)(d * rsqrtf(var + 1e-5f) * w[c] + bia[c]);
}

// ---------------- LayerNorm on row-major (M, 128) -> bf16 ----------------
__global__ void ln_kernel(const float* __restrict__ in,
                          const float* __restrict__ w,
                          const float* __restrict__ bia,
                          __bf16* __restrict__ outb) {
    __shared__ float s[128];
    int row = blockIdx.x;
    int c   = threadIdx.x;
    float v = in[(size_t)row * CDIM + c];
    float mean = block_sum128(v, s) * (1.0f / 128.0f);
    float d = v - mean;
    float var = block_sum128(d * d, s) * (1.0f / 128.0f);
    outb[(size_t)row * CDIM + c] = (__bf16)(d * rsqrtf(var + 1e-5f) * w[c] + bia[c]);
}

// ---------------- GEMM epilogue (bias / gelu / residual / store) ----------------
__device__ __forceinline__ void gemm_epilogue(v8f acc, int row0, int half, int n, int Nlog,
                                              void* __restrict__ Cout, int ldc,
                                              const float* __restrict__ bias,
                                              const float* __restrict__ resid, int ldr,
                                              int act, int tstore, int obf) {
    int  ncl = (n < Nlog) ? n : (Nlog - 1);
    bool nok = n < Nlog;
    float bv = bias ? bias[ncl] : 0.0f;
    #pragma unroll
    for (int r = 0; r < 8; ++r) {
        int m = row0 + r + 8 * half;     // C/D layout: VGPR r -> M = r + 8*half
        float v = acc[r] + bv;
        if (act == 1) v = 0.5f * v * (1.0f + erff(v * 0.70710678118f));
        if (resid) v += resid[(size_t)m * ldr + ncl];
        if (nok) {
            if (obf) {
                ((__bf16*)Cout)[(size_t)m * ldc + n] = (__bf16)v;
            } else if (!tstore) {
                ((float*)Cout)[(size_t)m * ldc + n] = v;
            } else {
                int b = m / LSEQ, l = m - b * LSEQ;
                ((float*)Cout)[((size_t)b * Nlog + n) * LSEQ + l] = v;
            }
        }
    }
}

// ---------------- WMMA bf16 GEMM: C[M,N] = A[M,K] * W[N,K]^T (+epilogue) ---------
// A and W are pre-converted bf16. One wave per 16x32 output tile (two 16x16 WMMA
// tiles sharing the A fragment). Fragments are contiguous bf16 runs gathered with
// 16-byte loads and assembled via shufflevector -> zero conversion VALU in the
// hot loop. W row pointers CLAMPED for out-of-range columns (D column n depends
// only on B column n); only the store is guarded, EXEC stays all-ones.
__global__ void gemm_bf16_wmma(const __bf16* __restrict__ A, int lda,
                               const __bf16* __restrict__ W, int ldw,
                               void* __restrict__ Cout, int ldc,
                               int K, int Nlog,
                               const float* __restrict__ bias,
                               const float* __restrict__ resid, int ldr,
                               int act, int tstore, int obf) {
    int row0 = blockIdx.x << 4;
    int n0   = blockIdx.y << 5;      // 32 output columns per wave
    int lane = threadIdx.x & 31;
    int half = lane >> 4;            // 0: lanes 0-15, 1: lanes 16-31
    int ml   = lane & 15;
    int nA   = n0 + ml;
    int nB   = n0 + 16 + ml;
    int nAc  = (nA < Nlog) ? nA : (Nlog - 1);
    int nBc  = (nB < Nlog) ? nB : (Nlog - 1);

    const __bf16* Arow  = A + (size_t)(row0 + ml) * lda;
    const __bf16* WrowA = W + (size_t)nAc * ldw;
    const __bf16* WrowB = W + (size_t)nBc * ldw;

    v8f acc0 = {}, acc1 = {};
    #pragma unroll 2
    for (int k0 = 0; k0 < K; k0 += 32) {
        // A 16x32 bf16, contiguous k order shared with B: half h holds K [k0+16h .. +15]
        v8bf a_lo = *(const v8bf*)(Arow + k0 + half * 16);
        v8bf a_hi = *(const v8bf*)(Arow + k0 + half * 16 + 8);
        // B 32x16 bf16, same k order as A: lane = N, half h holds K [k0+16h .. +15]
        v8bf b0lo = *(const v8bf*)(WrowA + k0 + half * 16);
        v8bf b0hi = *(const v8bf*)(WrowA + k0 + half * 16 + 8);
        v8bf b1lo = *(const v8bf*)(WrowB + k0 + half * 16);
        v8bf b1hi = *(const v8bf*)(WrowB + k0 + half * 16 + 8);
        v16bf af  = __builtin_shufflevector(a_lo, a_hi,
                        0,1,2,3,4,5,6,7,8,9,10,11,12,13,14,15);
        v16bf bf0 = __builtin_shufflevector(b0lo, b0hi,
                        0,1,2,3,4,5,6,7,8,9,10,11,12,13,14,15);
        v16bf bf1 = __builtin_shufflevector(b1lo, b1hi,
                        0,1,2,3,4,5,6,7,8,9,10,11,12,13,14,15);
        acc0 = __builtin_amdgcn_wmma_f32_16x16x32_bf16(
                   false, af, false, bf0, (short)0, acc0, false, false);
        acc1 = __builtin_amdgcn_wmma_f32_16x16x32_bf16(
                   false, af, false, bf1, (short)0, acc1, false, false);
    }

    gemm_epilogue(acc0, row0, half, nA, Nlog, Cout, ldc, bias, resid, ldr, act, tstore, obf);
    gemm_epilogue(acc1, row0, half, nB, Nlog, Cout, ldc, bias, resid, ldr, act, tstore, obf);
}

// ---------------- depthwise causal conv1d(4) + SiLU (f32 + bf16 copies) ---------
__global__ void conv_silu_kernel(const float* __restrict__ xz,   // (M, 512), xi = [:,0:256]
                                 const float* __restrict__ cw,   // (256, 4)
                                 const float* __restrict__ cb,   // (256)
                                 float* __restrict__ xc,         // (M, 256) f32 (scan)
                                 __bf16* __restrict__ xcb) {     // (M, 256) bf16 (GEMM)
    int t = blockIdx.x * blockDim.x + threadIdx.x;
    if (t >= MROWS * DINNER) return;
    int d   = t & (DINNER - 1);
    int row = t >> 8;
    int b = row / LSEQ, l = row - b * LSEQ;
    float acc = cb[d];
    #pragma unroll
    for (int j = 0; j < 4; ++j) {
        int ls = l + j - 3;
        if (ls >= 0)
            acc += cw[d * 4 + j] * xz[(size_t)(b * LSEQ + ls) * 512 + d];
    }
    float s = acc / (1.0f + expf(-acc));   // silu
    xc[t]  = s;
    xcb[t] = (__bf16)s;
}

// ---------------- dt_proj (K=8) + softplus ----------------
__global__ void dt_softplus_kernel(const float* __restrict__ xdbl,  // (M, 48) [0:8]=dt
                                   const float* __restrict__ dtw,   // (256, 8)
                                   const float* __restrict__ dtb,   // (256)
                                   float* __restrict__ delta) {     // (M, 256)
    int t = blockIdx.x * blockDim.x + threadIdx.x;
    if (t >= MROWS * DINNER) return;
    int d   = t & (DINNER - 1);
    int row = t >> 8;
    const float* dv = xdbl + (size_t)row * XDBL_LD;
    float acc = dtb[d];
    #pragma unroll
    for (int r = 0; r < DTRANK; ++r) acc += dv[r] * dtw[d * DTRANK + r];
    delta[t] = (acc > 20.0f) ? acc : log1pf(expf(acc));
}

// ---------------- selective scan + D skip + SiLU(z) gate -> bf16 ----------------
// 16 state lanes (n) per channel d in one shuffle group; wave32 = 2 channels.
// grid = B*16 blocks of 256 threads (16 channels x 16 states each).
__global__ void scan_kernel(const float* __restrict__ delta,  // (M, 256)
                            const float* __restrict__ xc,     // (M, 256)
                            const float* __restrict__ xdbl,   // (M, 48) [8:24]=B [24:40]=C
                            const float* __restrict__ xz,     // (M, 512), z = [:,256:512]
                            const float* __restrict__ A_log,  // (256, 16)
                            const float* __restrict__ Dvec,   // (256)
                            __bf16* __restrict__ ygb) {       // (M, 256) bf16 (GEMM)
    int b    = blockIdx.x >> 4;
    int dgrp = blockIdx.x & 15;
    int tid  = threadIdx.x;
    int d = dgrp * 16 + (tid >> 4);
    int n = tid & 15;
    float Aa = -expf(A_log[d * NSTATE + n]);
    float Dv = Dvec[d];
    float h = 0.0f;
    size_t rbase = (size_t)b * LSEQ;
    for (int l = 0; l < LSEQ; ++l) {
        size_t row = rbase + l;
        float dv = delta[row * DINNER + d];
        float u  = xc[row * DINNER + d];
        const float* db = xdbl + row * XDBL_LD;
        float Bv = db[DTRANK + n];
        float Cv = db[DTRANK + NSTATE + n];
        h = expf(dv * Aa) * h + dv * u * Bv;
        float y = h * Cv;
        #pragma unroll
        for (int off = 8; off > 0; off >>= 1) y += __shfl_xor(y, off, 16);
        if (n == 0) {
            float z = xz[row * 512 + DINNER + d];
            ygb[row * DINNER + d] = (__bf16)((y + u * Dv) * (z / (1.0f + expf(-z))));
        }
    }
}

// ---------------- launch ----------------
extern "C" void kernel_launch(void* const* d_in, const int* in_sizes, int n_in,
                              void* d_out, int out_size, void* d_ws, size_t ws_size,
                              hipStream_t stream) {
    const float* x         = (const float*)d_in[0];
    const float* norm1_w   = (const float*)d_in[1];
    const float* norm1_b   = (const float*)d_in[2];
    const float* norm2_w   = (const float*)d_in[3];
    const float* norm2_b   = (const float*)d_in[4];
    const float* in_proj_w = (const float*)d_in[5];   // (512, 128)
    const float* conv_w    = (const float*)d_in[6];   // (256, 4)
    const float* conv_b    = (const float*)d_in[7];
    const float* x_proj_w  = (const float*)d_in[8];   // (40, 256)
    const float* dt_proj_w = (const float*)d_in[9];   // (256, 8)
    const float* dt_proj_b = (const float*)d_in[10];
    const float* A_log     = (const float*)d_in[11];  // (256, 16)
    const float* Dvec      = (const float*)d_in[12];
    const float* out_proj_w= (const float*)d_in[13];  // (128, 256)
    const float* fc1_w     = (const float*)d_in[14];  // (512, 128)
    const float* fc1_b     = (const float*)d_in[15];
    const float* fc2_w     = (const float*)d_in[16];  // (128, 512)
    const float* fc2_b     = (const float*)d_in[17];
    float* out = (float*)d_out;

    // ---------- scratch layout: f32 region then bf16 region ----------
    float* wsf = (float*)d_ws;
    size_t of = 0;
    float* xf    = wsf + of; of += (size_t)MROWS * CDIM;     // (M,128)
    float* xz    = wsf + of; of += (size_t)MROWS * 512;      // (M,512)
    float* xc    = wsf + of; of += (size_t)MROWS * DINNER;   // (M,256)
    float* xdbl  = wsf + of; of += (size_t)MROWS * XDBL_LD;  // (M,48)
    float* delta = wsf + of; of += (size_t)MROWS * DINNER;   // (M,256)
    float* xm    = wsf + of; of += (size_t)MROWS * CDIM;     // (M,128)

    __bf16* wsb = (__bf16*)(wsf + of);
    size_t ob = 0;
    __bf16* h1b   = wsb + ob; ob += (size_t)MROWS * CDIM;    // (M,128)  reused as h2b
    __bf16* xcb   = wsb + ob; ob += (size_t)MROWS * DINNER;  // (M,256)
    __bf16* ygb   = wsb + ob; ob += (size_t)MROWS * DINNER;  // (M,256)
    __bf16* mlp1b = wsb + ob; ob += (size_t)MROWS * 512;     // (M,512)
    __bf16* w_inb = wsb + ob; ob += 512 * 128;
    __bf16* w_xpb = wsb + ob; ob += 40 * 256;
    __bf16* w_opb = wsb + ob; ob += 128 * 256;
    __bf16* w_f1b = wsb + ob; ob += 512 * 128;
    __bf16* w_f2b = wsb + ob; ob += 128 * 512;
    __bf16* h2b   = h1b;

    // 0) weights -> bf16 (tiny; every call, deterministic)
    f32_to_bf16_kernel<<<(512 * 128 + 255) / 256, 256, 0, stream>>>(in_proj_w, w_inb, 512 * 128);
    f32_to_bf16_kernel<<<(40 * 256 + 255) / 256, 256, 0, stream>>>(x_proj_w, w_xpb, 40 * 256);
    f32_to_bf16_kernel<<<(128 * 256 + 255) / 256, 256, 0, stream>>>(out_proj_w, w_opb, 128 * 256);
    f32_to_bf16_kernel<<<(512 * 128 + 255) / 256, 256, 0, stream>>>(fc1_w, w_f1b, 512 * 128);
    f32_to_bf16_kernel<<<(128 * 512 + 255) / 256, 256, 0, stream>>>(fc2_w, w_f2b, 128 * 512);

    // 1) transpose + LN1 -> xf (f32), h1b (bf16)
    transpose_ln_kernel<<<MROWS, 128, 0, stream>>>(x, norm1_w, norm1_b, xf, h1b);

    // 2) in_proj: (M,128) x (512,128)^T -> xz (M,512) f32
    gemm_bf16_wmma<<<dim3(MROWS / 16, 512 / 32), 32, 0, stream>>>(
        h1b, CDIM, w_inb, CDIM, xz, 512, CDIM, 512,
        nullptr, nullptr, 0, 0, 0, 0);

    // 3) depthwise causal conv + SiLU -> xc (f32), xcb (bf16)
    conv_silu_kernel<<<(MROWS * DINNER) / 256, 256, 0, stream>>>(xz, conv_w, conv_b, xc, xcb);

    // 4) x_proj: (M,256) x (40,256)^T -> xdbl (M,48 padded, 40 valid) f32
    gemm_bf16_wmma<<<dim3(MROWS / 16, 2), 32, 0, stream>>>(
        xcb, DINNER, w_xpb, DINNER, xdbl, XDBL_LD, DINNER, 40,
        nullptr, nullptr, 0, 0, 0, 0);

    // 5) dt_proj + softplus -> delta (f32)
    dt_softplus_kernel<<<(MROWS * DINNER) / 256, 256, 0, stream>>>(
        xdbl, dt_proj_w, dt_proj_b, delta);

    // 6) selective scan + D skip + SiLU(z) gate -> ygb (bf16)
    scan_kernel<<<BSZ * 16, 256, 0, stream>>>(delta, xc, xdbl, xz, A_log, Dvec, ygb);

    // 7) out_proj + residual(xf) -> xm (f32)
    gemm_bf16_wmma<<<dim3(MROWS / 16, CDIM / 32), 32, 0, stream>>>(
        ygb, DINNER, w_opb, DINNER, xm, CDIM, DINNER, CDIM,
        nullptr, xf, CDIM, 0, 0, 0);

    // 8) LN2 -> h2b (bf16)
    ln_kernel<<<MROWS, 128, 0, stream>>>(xm, norm2_w, norm2_b, h2b);

    // 9) fc1 + bias + exact GELU -> mlp1b (M,512) bf16
    gemm_bf16_wmma<<<dim3(MROWS / 16, 512 / 32), 32, 0, stream>>>(
        h2b, CDIM, w_f1b, CDIM, mlp1b, 512, CDIM, 512,
        fc1_b, nullptr, 0, 1, 0, 1);

    // 10) fc2 + bias + residual(xm), scatter-store into (B, C, L) f32 output
    gemm_bf16_wmma<<<dim3(MROWS / 16, CDIM / 32), 32, 0, stream>>>(
        mlp1b, 512, w_f2b, 512, out, CDIM, 512, CDIM,
        fc2_b, xm, CDIM, 0, 1, 0);
}
